// MultiLayerGATv2_1288490189686
// MI455X (gfx1250) — matrix operand order repaired and checked by reading, not compiled, not run on hardware
//
#include <hip/hip_runtime.h>
#include <hip/hip_bf16.h>

// ---------------------------------------------------------------------------
// Multi-layer GATv2 for gfx1250:
//   - Feature GEMMs via v_wmma_f32_16x16x32_bf16, weights staged in LDS via
//     gfx1250 async-to-LDS loads (ASYNCcnt) when the builtin is available
//   - Edge softmax via float-key atomicMax + atomicAdd (L2-resident xl/xr)
//   - Fused LayerNorm + ELU + residual + bf16 re-quantization per layer
// ---------------------------------------------------------------------------

typedef __attribute__((ext_vector_type(16))) __bf16 v16bf;
typedef __attribute__((ext_vector_type(8)))  float  v8f;
typedef __attribute__((ext_vector_type(4)))  int    v4i;

#define DHID 128
#define NHEAD 8
#define SLOPE 0.2f
#define LNEPS 1e-5f

#if defined(__has_builtin)
#if __has_builtin(__builtin_amdgcn_global_load_async_to_lds_b128) && \
    __has_builtin(__builtin_amdgcn_s_wait_asynccnt)
#define USE_ASYNC_LDS 1
#endif
#endif
#ifndef USE_ASYNC_LDS
#define USE_ASYNC_LDS 0
#endif

#if USE_ASYNC_LDS
typedef __attribute__((address_space(1))) v4i* gptr_v4i;   // global
typedef __attribute__((address_space(3))) v4i* lptr_v4i;   // LDS
#endif

struct Frag32 { uint4 lo, hi; };   // 32 bytes == v16bf

// A fragment (16x32 bf16): lane = M (&15), half = lane>>4.
// elements 0..7  -> K = kbase + half*8 + i
// elements 8..15 -> K = kbase + 16 + half*8 + (i-8)
__device__ __forceinline__ v16bf load_frag_a(const __bf16* row_kbase, int half) {
    Frag32 t;
    t.lo = *(const uint4*)(row_kbase + half * 8);
    t.hi = *(const uint4*)(row_kbase + 16 + half * 8);
    return __builtin_bit_cast(v16bf, t);
}

// Unused: 16 contiguous bf16 at wrow_n0 -> elements 0..15 (B frags use load_frag_a, lane&15 = N)
__device__ __forceinline__ v16bf load_frag_b(const __bf16* wrow_n0) {
    Frag32 t;
    t.lo = *(const uint4*)(wrow_n0);
    t.hi = *(const uint4*)(wrow_n0 + 8);
    return __builtin_bit_cast(v16bf, t);
}

// monotonic float <-> uint key for atomicMax-based segment max
__device__ __forceinline__ unsigned fkey(float f) {
    unsigned u = __float_as_uint(f);
    return (u & 0x80000000u) ? ~u : (u | 0x80000000u);
}
__device__ __forceinline__ float fdec(unsigned k) {
    return (k & 0x80000000u) ? __uint_as_float(k & 0x7fffffffu)
                             : __uint_as_float(~k);
}

// ---------------------------------------------------------------------------
// Weight convert: fp32 row-major W[k][n] -> bf16 transposed W^T[n][k] (B frags
// read K-contiguous bf16 per N-row, same layout as A frags).
__global__ void k_convert_weights(const float* __restrict__ Wl,
                                  const float* __restrict__ Wr,
                                  __bf16* __restrict__ wbf, int total) {
    int i = blockIdx.x * blockDim.x + threadIdx.x;
    if (i >= total) return;
    int per = 2 * DHID * DHID;
    int l = i / per, r = i % per;
    int sel = r / (DHID * DHID), j = r % (DHID * DHID);
    const float* src = sel ? Wr : Wl;
    wbf[l * per + sel * DHID * DHID + (j % DHID) * DHID + j / DHID] = (__bf16)src[l * DHID * DHID + j];
}

// h = x (residual stream lives in d_out), plus bf16 mirror for the GEMM
__global__ void k_init_h(const float* __restrict__ x, float* __restrict__ h,
                         __bf16* __restrict__ hb, int total) {
    int i = blockIdx.x * blockDim.x + threadIdx.x;
    if (i >= total) return;
    float v = x[i];
    h[i] = v;
    hb[i] = (__bf16)v;
}

// zero out-accumulator, segment-max keys, softmax denominators
__global__ void k_layer_init(float* __restrict__ outb, unsigned* __restrict__ mkey,
                             float* __restrict__ denom, int N) {
    int i = blockIdx.x * blockDim.x + threadIdx.x;
    if (i < N * DHID) outb[i] = 0.0f;
    if (i < N * NHEAD) { mkey[i] = 0u; denom[i] = 0.0f; }
}

// ---------------------------------------------------------------------------
// Per-wave GEMM body.  GUARD=false is the hot path (whole 16-row tile valid):
// stores are unpredicated, no exec-mask ladders.
template <bool GUARD>
__device__ __forceinline__ void gemm_body(const __bf16* wsm, int m0, int half,
                                          int mlo, int lane, v16bf a0, v16bf a1,
                                          v16bf a2, v16bf a3, float* __restrict__ xl,
                                          float* __restrict__ xr, int N) {
    for (int sel = 0; sel < 2; ++sel) {
        const __bf16* wp = wsm + sel * DHID * DHID;   // LDS
        float* x = sel ? xr : xl;
        for (int nt = 0; nt < 8; ++nt) {
            int n0 = nt * 16;
            // issue all four B frags (8x ds_load_b128), then chain 4 WMMAs
            v16bf b0 = load_frag_a(wp + (size_t)(n0 + mlo) * DHID + 0 * 32, half);
            v16bf b1 = load_frag_a(wp + (size_t)(n0 + mlo) * DHID + 1 * 32, half);
            v16bf b2 = load_frag_a(wp + (size_t)(n0 + mlo) * DHID + 2 * 32, half);
            v16bf b3 = load_frag_a(wp + (size_t)(n0 + mlo) * DHID + 3 * 32, half);
            v8f c = {};
            c = __builtin_amdgcn_wmma_f32_16x16x32_bf16(false, a0, false, b0, (short)0, c, false, false);
            c = __builtin_amdgcn_wmma_f32_16x16x32_bf16(false, a1, false, b1, (short)0, c, false, false);
            c = __builtin_amdgcn_wmma_f32_16x16x32_bf16(false, a2, false, b2, (short)0, c, false, false);
            c = __builtin_amdgcn_wmma_f32_16x16x32_bf16(false, a3, false, b3, (short)0, c, false, false);
            // D layout: element j -> row m0 + j + 8*half, col n0 + mlo
            float* xb = x + (size_t)(m0 + 8 * half) * DHID + n0 + mlo;
#pragma unroll
            for (int j = 0; j < 8; ++j) {
                if (!GUARD || (m0 + 8 * half + j < N)) xb[(size_t)j * DHID] = c[j];
            }
        }
    }
}

// Dual GEMM: xl = h @ Wl, xr = h @ Wr.  Both weights (64KB bf16) staged into
// LDS once per block via gfx1250 async-to-LDS loads; one wave per 16-row
// strip, A frags (full K=128) reused across all 16 output column tiles.
__global__ void __launch_bounds__(256) k_gemm_dual(
        const __bf16* __restrict__ hb, const __bf16* __restrict__ wpair,
        float* __restrict__ xl, float* __restrict__ xr, int nTiles, int N) {
    __shared__ __bf16 wsm[2 * DHID * DHID];   // 64 KB of 320 KB WGP LDS

    // --- stage Wl|Wr into LDS: 256 threads x 16 chunks x 16B = 64KB ---
    {
        int tid = threadIdx.x;
#pragma unroll
        for (int c = 0; c < 16; ++c) {
            int elt = (c * 256 + tid) * 8;    // 8 bf16 = 16 bytes per op
#if USE_ASYNC_LDS
            __builtin_amdgcn_global_load_async_to_lds_b128(
                (gptr_v4i)(wpair + elt), (lptr_v4i)(&wsm[elt]), 0, 0);
#else
            *(uint4*)&wsm[elt] = *(const uint4*)(wpair + elt);
#endif
        }
#if USE_ASYNC_LDS
        __builtin_amdgcn_s_wait_asynccnt(0);
#endif
        __syncthreads();
    }

    int wave = threadIdx.x >> 5, lane = threadIdx.x & 31;
    int tile = blockIdx.x * (blockDim.x >> 5) + wave;
    if (tile >= nTiles) return;
    int half = lane >> 4, mlo = lane & 15;
    int m0 = tile * 16;
    int mrow = m0 + mlo; if (mrow >= N) mrow = N - 1;

    const __bf16* arow = hb + (size_t)mrow * DHID;
    v16bf a0 = load_frag_a(arow + 0,  half);
    v16bf a1 = load_frag_a(arow + 32, half);
    v16bf a2 = load_frag_a(arow + 64, half);
    v16bf a3 = load_frag_a(arow + 96, half);

    if (m0 + 16 <= N) {
        gemm_body<false>(wsm, m0, half, mlo, lane, a0, a1, a2, a3, xl, xr, N);
    } else {
        gemm_body<true>(wsm, m0, half, mlo, lane, a0, a1, a2, a3, xl, xr, N);
    }
}

// ---------------------------------------------------------------------------
// Pass 1: per-edge logits + segment max.  One wave per edge: each lane owns
// 4 channels, heads are 4-lane groups (C=16).
__global__ void k_edge_logits(const float* __restrict__ xl, const float* __restrict__ xr,
                              const int* __restrict__ ei, const float* __restrict__ att,
                              float* __restrict__ exlog, unsigned* __restrict__ mkey,
                              int E, int ETOT) {
    int gid = blockIdx.x * (blockDim.x >> 5) + (threadIdx.x >> 5);
    if (gid >= ETOT) return;
    int lane = threadIdx.x & 31;
    int s, d;
    if (gid < E) { s = ei[gid]; d = ei[E + gid]; } else { s = d = gid - E; }

    float4 a = *(const float4*)(xl + (size_t)s * DHID + lane * 4);
    float4 b = *(const float4*)(xr + (size_t)d * DHID + lane * 4);
    float4 t = *(const float4*)(att + lane * 4);
    float p = 0.0f, v;
    v = a.x + b.x; v = v > 0.0f ? v : SLOPE * v; p += v * t.x;
    v = a.y + b.y; v = v > 0.0f ? v : SLOPE * v; p += v * t.y;
    v = a.z + b.z; v = v > 0.0f ? v : SLOPE * v; p += v * t.z;
    v = a.w + b.w; v = v > 0.0f ? v : SLOPE * v; p += v * t.w;
    p += __shfl_xor(p, 1);
    p += __shfl_xor(p, 2);
    if ((lane & 3) == 0) {
        int h = lane >> 2;
        exlog[(size_t)gid * NHEAD + h] = p;
        atomicMax(mkey + (size_t)d * NHEAD + h, fkey(p));
    }
}

// Pass 2: ex = exp(logit - max); accumulate denominators. Thread per (edge,head).
__global__ void k_edge_exp(const int* __restrict__ ei, float* __restrict__ exlog,
                           const unsigned* __restrict__ mkey, float* __restrict__ denom,
                           int E, int ETOT) {
    long long tid = (long long)blockIdx.x * blockDim.x + threadIdx.x;
    if (tid >= (long long)ETOT * NHEAD) return;
    int e = (int)(tid >> 3), h = (int)(tid & 7);
    int d = (e < E) ? ei[E + e] : e - E;
    float m = fdec(mkey[(size_t)d * NHEAD + h]);
    float ex = __expf(exlog[tid] - m);
    exlog[tid] = ex;
    atomicAdd(denom + (size_t)d * NHEAD + h, ex);
}

// Pass 3: out[d] += alpha * xl[s].  Wave per edge, 4 channels per lane.
__global__ void k_edge_scatter(const float* __restrict__ xl, const int* __restrict__ ei,
                               const float* __restrict__ exlog, const float* __restrict__ denom,
                               float* __restrict__ outb, int E, int ETOT) {
    int gid = blockIdx.x * (blockDim.x >> 5) + (threadIdx.x >> 5);
    if (gid >= ETOT) return;
    int lane = threadIdx.x & 31;
    int s, d;
    if (gid < E) { s = ei[gid]; d = ei[E + gid]; } else { s = d = gid - E; }
    int h = lane >> 2;
    float alpha = exlog[(size_t)gid * NHEAD + h] / denom[(size_t)d * NHEAD + h];
    float4 v = *(const float4*)(xl + (size_t)s * DHID + lane * 4);
    float* ob = outb + (size_t)d * DHID + lane * 4;
    atomicAdd(ob + 0, alpha * v.x);
    atomicAdd(ob + 1, alpha * v.y);
    atomicAdd(ob + 2, alpha * v.z);
    atomicAdd(ob + 3, alpha * v.w);
}

// ---------------------------------------------------------------------------
// Fused: +bias, LayerNorm, ELU, residual, and bf16 re-quantization for next
// layer's GEMM.  Wave per node, full-wave shuffle reductions.
__global__ void k_finalize(const float* __restrict__ outb, const float* __restrict__ bias,
                           const float* __restrict__ gamma, const float* __restrict__ beta,
                           float* __restrict__ h, __bf16* __restrict__ hb, int N) {
    int gid = blockIdx.x * (blockDim.x >> 5) + (threadIdx.x >> 5);
    if (gid >= N) return;
    int lane = threadIdx.x & 31;
    size_t base = (size_t)gid * DHID + lane * 4;

    float4 y  = *(const float4*)(outb + base);
    float4 bi = *(const float4*)(bias + lane * 4);
    y.x += bi.x; y.y += bi.y; y.z += bi.z; y.w += bi.w;

    float s = y.x + y.y + y.z + y.w;
#pragma unroll
    for (int o = 1; o < 32; o <<= 1) s += __shfl_xor(s, o);
    float mu = s * (1.0f / DHID);

    float dx = y.x - mu, dy = y.y - mu, dz = y.z - mu, dw = y.w - mu;
    float q = dx * dx + dy * dy + dz * dz + dw * dw;
#pragma unroll
    for (int o = 1; o < 32; o <<= 1) q += __shfl_xor(q, o);
    float rs = rsqrtf(q * (1.0f / DHID) + LNEPS);

    float4 g = *(const float4*)(gamma + lane * 4);
    float4 be = *(const float4*)(beta + lane * 4);
    float z0 = dx * rs * g.x + be.x;
    float z1 = dy * rs * g.y + be.y;
    float z2 = dz * rs * g.z + be.z;
    float z3 = dw * rs * g.w + be.w;
    float e0 = z0 > 0.0f ? z0 : __expf(z0) - 1.0f;
    float e1 = z1 > 0.0f ? z1 : __expf(z1) - 1.0f;
    float e2 = z2 > 0.0f ? z2 : __expf(z2) - 1.0f;
    float e3 = z3 > 0.0f ? z3 : __expf(z3) - 1.0f;

    float4 ho = *(const float4*)(h + base);
    float h0 = ho.x + e0, h1 = ho.y + e1, h2 = ho.z + e2, h3 = ho.w + e3;
    float4 hn = {h0, h1, h2, h3};
    *(float4*)(h + base) = hn;
    hb[base + 0] = (__bf16)h0;
    hb[base + 1] = (__bf16)h1;
    hb[base + 2] = (__bf16)h2;
    hb[base + 3] = (__bf16)h3;
}

// ---------------------------------------------------------------------------
extern "C" void kernel_launch(void* const* d_in, const int* in_sizes, int n_in,
                              void* d_out, int out_size, void* d_ws, size_t ws_size,
                              hipStream_t stream) {
    const float* x     = (const float*)d_in[0];
    const float* Wl    = (const float*)d_in[1];
    const float* Wr    = (const float*)d_in[2];
    const float* att   = (const float*)d_in[3];
    const float* bias  = (const float*)d_in[4];
    const float* gamma = (const float*)d_in[5];
    const float* beta  = (const float*)d_in[6];
    const int*   ei    = (const int*)d_in[7];

    const int N = in_sizes[0] / DHID;
    const int E = in_sizes[7] / 2;
    const int ETOT = E + N;
    const int L = 3;

    char* ws = (char*)d_ws;
    size_t off = 0;
    auto take = [&](size_t bytes) -> char* {
        char* p = ws + off;
        off = (off + bytes + 255) & ~(size_t)255;
        return p;
    };
    __bf16*   hb    = (__bf16*)take((size_t)N * DHID * sizeof(__bf16));
    __bf16*   wbf   = (__bf16*)take((size_t)L * 2 * DHID * DHID * sizeof(__bf16));
    float*    xl    = (float*)take((size_t)N * DHID * sizeof(float));
    float*    xr    = (float*)take((size_t)N * DHID * sizeof(float));
    float*    exlog = (float*)take((size_t)ETOT * NHEAD * sizeof(float));
    unsigned* mkey  = (unsigned*)take((size_t)N * NHEAD * sizeof(unsigned));
    float*    denom = (float*)take((size_t)N * NHEAD * sizeof(float));
    float*    outb  = (float*)take((size_t)N * DHID * sizeof(float));
    float*    h     = (float*)d_out;   // residual stream lives in d_out

    dim3 blk(256);
    const int wtotal = L * 2 * DHID * DHID;
    k_convert_weights<<<(wtotal + 255) / 256, blk, 0, stream>>>(Wl, Wr, wbf, wtotal);
    k_init_h<<<(N * DHID + 255) / 256, blk, 0, stream>>>(x, h, hb, N * DHID);

    const int nTiles = (N + 15) / 16;
    for (int l = 0; l < L; ++l) {
        const __bf16* wpair = wbf + (size_t)l * 2 * DHID * DHID;  // Wl|Wr contiguous
        k_layer_init<<<(N * DHID + 255) / 256, blk, 0, stream>>>(outb, mkey, denom, N);
        k_gemm_dual<<<(nTiles + 7) / 8, blk, 0, stream>>>(hb, wpair, xl, xr, nTiles, N);
        k_edge_logits<<<(ETOT + 7) / 8, blk, 0, stream>>>(xl, xr, ei, att + l * DHID,
                                                          exlog, mkey, E, ETOT);
        k_edge_exp<<<(int)(((long long)ETOT * NHEAD + 255) / 256), blk, 0, stream>>>(
            ei, exlog, mkey, denom, E, ETOT);
        k_edge_scatter<<<(ETOT + 7) / 8, blk, 0, stream>>>(xl, ei, exlog, denom, outb, E, ETOT);
        k_finalize<<<(N + 7) / 8, blk, 0, stream>>>(outb, bias + l * DHID, gamma + l * DHID,
                                                    beta + l * DHID, h, hb, N);
    }
}
